// Attention_15607911153821
// MI455X (gfx1250) — matrix operand order repaired and checked by reading, not compiled, not run on hardware
//
#include <hip/hip_runtime.h>

typedef __attribute__((ext_vector_type(16))) _Float16 v16h;
typedef __attribute__((ext_vector_type(8)))  float    v8f;

#define SQ 2048
#define DH 64
#define NW 8               // waves per workgroup (256 threads)
#define QW 32              // queries per wave = two 16-row WMMA M-subtiles
#define QB (NW * QW)       // 256 queries per workgroup
#define KT 32              // key-tile (one WMMA K-step for PV)

__device__ __forceinline__ float fexp2(float x) { return __builtin_amdgcn_exp2f(x); }

__global__ __launch_bounds__(256) void fa_fwd_f16wmma(
    const float* __restrict__ Q,
    const float* __restrict__ K,
    const float* __restrict__ V,
    float* __restrict__ O)
{
    __shared__ _Float16 sK [2][KT][DH + 8];   // double-buffered keys (f16)
    __shared__ _Float16 sVt[2][DH][KT + 8];   // double-buffered V^T [dim][key]
    __shared__ _Float16 sP [NW][16][KT];      // per-wave C->A relayout scratch

    const int lane = threadIdx.x & 31;
    const int wave = threadIdx.x >> 5;
    const int n    = lane & 15;
    const int hi   = lane >> 4;

    const int    bh   = blockIdx.y;
    const int    qblk = blockIdx.x * QB;
    const int    q0   = qblk + wave * QW;
    // scalar (SGPR) copy of the wave-uniform row base -> uniform branches,
    // EXEC stays all-ones around WMMA blocks.
    const int    q0s  = __builtin_amdgcn_readfirstlane(q0);
    const size_t base = (size_t)bh * SQ * DH;

    // softmax in base-2 domain: fold 1/sqrt(D) * log2(e) into Q
    const float scale = 0.125f * 1.44269504088896f;

    v16h aQ[2][2];
    #pragma unroll
    for (int qs = 0; qs < 2; ++qs) {
        const float* q = Q + base + (size_t)(q0 + qs * 16 + n) * DH;
        #pragma unroll
        for (int c = 0; c < 2; ++c) {
            #pragma unroll
            for (int j = 0; j < 8; ++j) {
                aQ[qs][c][j]     = (_Float16)(q[c * 32 + 16 * hi +     j] * scale);
                aQ[qs][c][j + 8] = (_Float16)(q[c * 32 + 16 * hi + 8 + j] * scale);
            }
        }
    }

    v8f o[2][4] = {{{}, {}, {}, {}}, {{}, {}, {}, {}}};
    v8f lacc[2] = {{}, {}};
    float mrow[2][8];
    #pragma unroll
    for (int qs = 0; qs < 2; ++qs)
        #pragma unroll
        for (int r = 0; r < 8; ++r) mrow[qs][r] = -__builtin_inff();

    v16h onesB;
    #pragma unroll
    for (int j = 0; j < 16; ++j) onesB[j] = (_Float16)1.0f;

    const int nkt = (qblk + QB) / KT;         // causal: keys 0 .. qblk+QB-1

    // staging (256 thr): K = 32 rows x 8 segs of 8; V = 16 key-pairs x 16 segs of 4
    const int krow = threadIdx.x >> 3;
    const int kseg = (threadIdx.x & 7) * 8;
    const int vkp  = threadIdx.x >> 4;
    const int vseg = (threadIdx.x & 15) * 4;

    const float* kbase  = K + base + (size_t)krow * DH + kseg;
    const float* vbase0 = V + base + (size_t)(2 * vkp) * DH + vseg;
    const float* vbase1 = vbase0 + DH;

    {   // ---- prologue: stage tile 0 into buffer 0 ------------------------
        #pragma unroll
        for (int j = 0; j < 4; ++j) {
            union { _Float16 h[2]; unsigned u; } pk;
            pk.h[0] = (_Float16)kbase[2 * j]; pk.h[1] = (_Float16)kbase[2 * j + 1];
            *(unsigned*)&sK[0][krow][kseg + 2 * j] = pk.u;
        }
        #pragma unroll
        for (int j = 0; j < 4; ++j) {
            union { _Float16 h[2]; unsigned u; } pk;
            pk.h[0] = (_Float16)vbase0[j]; pk.h[1] = (_Float16)vbase1[j];
            *(unsigned*)&sVt[0][vseg + j][2 * vkp] = pk.u;
        }
    }
    __syncthreads();

    for (int kt = 0; kt < nkt; ++kt) {
        const int  cur      = kt & 1;
        const int  k0       = kt * KT;
        const bool haveNext = (kt + 1 < nkt);

        // ---- issue next tile's global loads early (latency hidden) -------
        float kf[8], vf0[4], vf1[4];
        if (haveNext) {
            const float* kp  = kbase  + (size_t)(kt + 1) * KT * DH;
            const float* vp0 = vbase0 + (size_t)(kt + 1) * KT * DH;
            const float* vp1 = vbase1 + (size_t)(kt + 1) * KT * DH;
            #pragma unroll
            for (int j = 0; j < 8; ++j) kf[j] = kp[j];
            #pragma unroll
            for (int j = 0; j < 4; ++j) { vf0[j] = vp0[j]; vf1[j] = vp1[j]; }
            if (kt + 2 < nkt) {            // global_prefetch_b8: warm L2
                __builtin_prefetch(kp  + KT * DH, 0, 1);
                __builtin_prefetch(vp0 + KT * DH, 0, 1);
            }
        }

        // ---- uniform (scalar) skip of fully-masked tiles -----------------
        if (k0 <= q0s + QW - 1) {
            v16h bk[2][2];
            #pragma unroll
            for (int nt = 0; nt < 2; ++nt) {
                const int key = nt * 16 + n;
                #pragma unroll
                for (int j = 0; j < 16; ++j) {
                    bk[nt][0][j] = sK[cur][key][     16 * hi + j];
                    bk[nt][1][j] = sK[cur][key][32 + 16 * hi + j];
                }
            }

            v8f s[2][2];
            #pragma unroll
            for (int qs = 0; qs < 2; ++qs)
                #pragma unroll
                for (int nt = 0; nt < 2; ++nt) {
                    v8f acc = {};
                    acc = __builtin_amdgcn_wmma_f32_16x16x32_f16(
                              false, aQ[qs][0], false, bk[nt][0], (short)0, acc, false, false);
                    acc = __builtin_amdgcn_wmma_f32_16x16x32_f16(
                              false, aQ[qs][1], false, bk[nt][1], (short)0, acc, false, false);
                    s[qs][nt] = acc;
                }

            v16h bv[4];
            #pragma unroll
            for (int dt = 0; dt < 4; ++dt) {
                const int d = dt * 16 + n;
                #pragma unroll
                for (int j = 0; j < 16; ++j)
                    bv[dt][j] = sVt[cur][d][16 * hi + j];
            }

            // uniform: does this tile cross the diagonal?
            if (k0 + KT - 1 > q0s) {
                #pragma unroll
                for (int qs = 0; qs < 2; ++qs)
                    #pragma unroll
                    for (int r = 0; r < 8; ++r) {
                        const int qrow = q0 + qs * 16 + r + 8 * hi;
                        if (k0 + n      > qrow) s[qs][0][r] = -__builtin_inff();
                        if (k0 + 16 + n > qrow) s[qs][1][r] = -__builtin_inff();
                    }
            }

            #pragma unroll
            for (int qs = 0; qs < 2; ++qs) {
                #pragma unroll
                for (int r = 0; r < 8; ++r) {
                    float mx = fmaxf(s[qs][0][r], s[qs][1][r]);
                    #pragma unroll
                    for (int msk = 1; msk < 16; msk <<= 1)
                        mx = fmaxf(mx, __shfl_xor(mx, msk, 32));

                    const float mnew = fmaxf(mrow[qs][r], mx);
                    const float corr = fexp2(mrow[qs][r] - mnew);   // v_exp_f32
                    mrow[qs][r] = mnew;

                    s[qs][0][r] = fexp2(s[qs][0][r] - mnew);
                    s[qs][1][r] = fexp2(s[qs][1][r] - mnew);

                    lacc[qs][r] *= corr;
                    #pragma unroll
                    for (int dt = 0; dt < 4; ++dt)
                        o[qs][dt][r] *= corr;
                }

                #pragma unroll
                for (int r = 0; r < 8; ++r) {
                    sP[wave][r + 8 * hi][n]      = (_Float16)s[qs][0][r];
                    sP[wave][r + 8 * hi][16 + n] = (_Float16)s[qs][1][r];
                }
                v16h aP;
                #pragma unroll
                for (int j = 0; j < 8; ++j) {
                    aP[j]     = sP[wave][n][16 * hi +     j];
                    aP[j + 8] = sP[wave][n][16 * hi + 8 + j];
                }

                #pragma unroll
                for (int dt = 0; dt < 4; ++dt)
                    o[qs][dt] = __builtin_amdgcn_wmma_f32_16x16x32_f16(
                                    false, aP, false, bv[dt], (short)0, o[qs][dt], false, false);
                lacc[qs] = __builtin_amdgcn_wmma_f32_16x16x32_f16(
                               false, aP, false, onesB, (short)0, lacc[qs], false, false);
            }
        }

        // ---- convert + store next tile into the alternate buffer ---------
        if (haveNext) {
            const int nxt = cur ^ 1;
            #pragma unroll
            for (int j = 0; j < 4; ++j) {
                union { _Float16 h[2]; unsigned u; } pk;
                pk.h[0] = (_Float16)kf[2 * j]; pk.h[1] = (_Float16)kf[2 * j + 1];
                *(unsigned*)&sK[nxt][krow][kseg + 2 * j] = pk.u;
            }
            #pragma unroll
            for (int j = 0; j < 4; ++j) {
                union { _Float16 h[2]; unsigned u; } pk;
                pk.h[0] = (_Float16)vf0[j]; pk.h[1] = (_Float16)vf1[j];
                *(unsigned*)&sVt[nxt][vseg + j][2 * vkp] = pk.u;
            }
        }
        __syncthreads();
    }

    // ---- normalize + coalesced stores -------------------------------------
    #pragma unroll
    for (int qs = 0; qs < 2; ++qs)
        #pragma unroll
        for (int r = 0; r < 8; ++r) {
            const float  inv = __builtin_amdgcn_rcpf(lacc[qs][r]);  // v_rcp_f32
            const size_t row = base + (size_t)(q0 + qs * 16 + r + 8 * hi) * DH;
            #pragma unroll
            for (int dt = 0; dt < 4; ++dt)
                O[row + dt * 16 + n] = o[qs][dt][r] * inv;
        }
}

extern "C" void kernel_launch(void* const* d_in, const int* in_sizes, int n_in,
                              void* d_out, int out_size, void* d_ws, size_t ws_size,
                              hipStream_t stream) {
    const float* Q = (const float*)d_in[0];
    const float* K = (const float*)d_in[1];
    const float* V = (const float*)d_in[2];
    // d_in[3] is the causal mask; causality is applied analytically.
    float* O = (float*)d_out;

    const int B = 4, H = 16;
    dim3 grid(SQ / QB, B * H);   // (8, 64)
    dim3 block(256);
    fa_fwd_f16wmma<<<grid, block, 0, stream>>>(Q, K, V, O);
}
